// Attention_70282844832541
// MI455X (gfx1250) — compile-verified, matrix operand order checked by reading, not run on hardware
//
#include <hip/hip_runtime.h>
#include <hip/hip_bf16.h>

// MI455X / gfx1250, wave32. f16-in / f32-acc WMMA everywhere.
typedef __attribute__((ext_vector_type(16))) _Float16 v16h;
typedef __attribute__((ext_vector_type(8)))  _Float16 v8h;
typedef __attribute__((ext_vector_type(8)))  float    v8f;

union V16U { v16h v; v8h h[2]; };

// Load a 16-lane-striped fragment: per-lane halves are K = {hi8..hi8+7, 16+hi8..16+hi8+7}
// p points at the start of this lane's row (A: row m=lane&15 of A; B: row n=lane&15 of W).
__device__ __forceinline__ v16h ld_frag(const _Float16* p, int hi8) {
  V16U u;
  u.h[0] = *(const v8h*)(p + hi8);
  u.h[1] = *(const v8h*)(p + hi8 + 16);
  return u.v;
}

__device__ __forceinline__ v8f wmma32(v16h a, v16h b, v8f c) {
  return __builtin_amdgcn_wmma_f32_16x16x32_f16(false, a, false, b, (short)0, c, false, false);
}

// ---------------------------------------------------------------- elementwise
__global__ void cvt_f32_f16(const float* __restrict__ in, _Float16* __restrict__ out, size_t n) {
  size_t i = (size_t)blockIdx.x * blockDim.x + threadIdx.x;
  if (i < n) out[i] = (_Float16)in[i];
}

// qf: [B*L, heads*64] f32 -> out [B, heads, L, 64] f16, with RoPE (cos/sin: [L,64])
__global__ void rope_pack(const float* __restrict__ qf, const float* __restrict__ cosp,
                          const float* __restrict__ sinp, _Float16* __restrict__ outh,
                          int heads, int applyRope) {
  constexpr int L = 2048;
  long long idx = (long long)blockIdx.x * blockDim.x + threadIdx.x;
  long long total = (long long)4 * L * heads * 64;
  if (idx >= total) return;
  int d = (int)(idx & 63);
  long long t = idx >> 6;
  int l = (int)(t % L);
  long long t2 = t / L;
  int h = (int)(t2 % heads);
  int b = (int)(t2 / heads);
  int ld = heads * 64;
  size_t row = (size_t)b * L + l;
  float v = qf[row * ld + h * 64 + d];
  if (applyRope) {
    float c  = cosp[(size_t)l * 64 + d];
    float sn = sinp[(size_t)l * 64 + d];
    float other = (d < 32) ? -qf[row * ld + h * 64 + d + 32]
                           :  qf[row * ld + h * 64 + d - 32];
    v = v * c + other * sn;
  }
  outh[(((size_t)b * heads + h) * L + l) * 64 + d] = (_Float16)v;
}

// vf: [B*L, 128] f32 -> vT [B, 2, 64, L] f16 (head-dim major so PV B-fragments are row reads)
__global__ void pack_vT(const float* __restrict__ vf, _Float16* __restrict__ vt) {
  constexpr int L = 2048, NKV = 2;
  long long idx = (long long)blockIdx.x * blockDim.x + threadIdx.x;
  long long total = (long long)4 * L * NKV * 64;
  if (idx >= total) return;
  int d = (int)(idx & 63);
  long long t = idx >> 6;
  int l = (int)(t % L);
  long long t2 = t / L;
  int h = (int)(t2 % NKV);
  int b = (int)(t2 / NKV);
  vt[(((size_t)b * NKV + h) * 64 + d) * L + l] =
      (_Float16)vf[((size_t)b * L + l) * 128 + h * 64 + d];
}

// ---------------------------------------------------------------- GEMM  C = A * W^T + bias
// A:[M,K] f16 row-major, W:[N,K] f16 row-major, C:[M,N] f32.
// One wave = 32x64 tile (2 A-frags x 4 B-frags -> 8 WMMA per 6 fragment loads),
// software-pipelined so next k-chunk's loads are outstanding during the WMMAs.
__global__ __launch_bounds__(128)
void gemm_xwt_wmma(const _Float16* __restrict__ A, const _Float16* __restrict__ W,
                   const float* __restrict__ bias, float* __restrict__ C,
                   int M, int N, int K) {
  int lane = threadIdx.x & 31;
  int wave = threadIdx.x >> 5;
  int tile = blockIdx.x * (blockDim.x >> 5) + wave;
  int ntiles = N >> 6, mtiles = M >> 5;
  if (tile >= ntiles * mtiles) return;
  int mt = tile / ntiles, nt = tile % ntiles;
  int m   = lane & 15;
  int hi  = lane >> 4;
  int hi8 = hi << 3;
  int n0  = nt * 64;

  const _Float16* arow0 = A + (size_t)(mt * 32 + m) * K;
  const _Float16* arow1 = arow0 + (size_t)16 * K;
  const _Float16* brow0 = W + (size_t)(n0 +  0 + m) * K;
  const _Float16* brow1 = W + (size_t)(n0 + 16 + m) * K;
  const _Float16* brow2 = W + (size_t)(n0 + 32 + m) * K;
  const _Float16* brow3 = W + (size_t)(n0 + 48 + m) * K;

  v8f acc[8] = {};   // acc[s*2+half]: half 0 -> rows 0..15 of tile, half 1 -> rows 16..31

  // prologue: fragments for k-chunk 0
  v16h a0 = ld_frag(arow0, hi8), a1 = ld_frag(arow1, hi8);
  v16h b0 = ld_frag(brow0, hi8), b1 = ld_frag(brow1, hi8);
  v16h b2 = ld_frag(brow2, hi8), b3 = ld_frag(brow3, hi8);

  for (int k0 = 32; k0 < K; k0 += 32) {
    // issue next chunk's loads first; WMMAs below depend only on previous regs
    v16h na0 = ld_frag(arow0 + k0, hi8), na1 = ld_frag(arow1 + k0, hi8);
    v16h nb0 = ld_frag(brow0 + k0, hi8), nb1 = ld_frag(brow1 + k0, hi8);
    v16h nb2 = ld_frag(brow2 + k0, hi8), nb3 = ld_frag(brow3 + k0, hi8);

    acc[0] = wmma32(a0, b0, acc[0]);  acc[1] = wmma32(a1, b0, acc[1]);
    acc[2] = wmma32(a0, b1, acc[2]);  acc[3] = wmma32(a1, b1, acc[3]);
    acc[4] = wmma32(a0, b2, acc[4]);  acc[5] = wmma32(a1, b2, acc[5]);
    acc[6] = wmma32(a0, b3, acc[6]);  acc[7] = wmma32(a1, b3, acc[7]);

    a0 = na0; a1 = na1; b0 = nb0; b1 = nb1; b2 = nb2; b3 = nb3;
  }
  // epilogue chunk
  acc[0] = wmma32(a0, b0, acc[0]);  acc[1] = wmma32(a1, b0, acc[1]);
  acc[2] = wmma32(a0, b1, acc[2]);  acc[3] = wmma32(a1, b1, acc[3]);
  acc[4] = wmma32(a0, b2, acc[4]);  acc[5] = wmma32(a1, b2, acc[5]);
  acc[6] = wmma32(a0, b3, acc[6]);  acc[7] = wmma32(a1, b3, acc[7]);

  int row0 = mt * 32;
#pragma unroll
  for (int s = 0; s < 4; ++s) {
    int col = n0 + s * 16 + m;
    float bv = bias ? bias[col] : 0.0f;
#pragma unroll
    for (int half = 0; half < 2; ++half) {
#pragma unroll
      for (int r = 0; r < 8; ++r)
        C[(size_t)(row0 + half * 16 + r + 8 * hi) * N + col] = acc[s * 2 + half][r] + bv;
    }
  }
}

// ---------------------------------------------------------------- flash attention
// qh:[B,NH,L,64] f16, kh:[B,NKV,L,64] f16, vT:[B,NKV,64,L] f16 -> attn_h:[B*L, NH*64] f16
// One wave per (b, h, 16-row q tile); causal mask computed analytically.
__global__ __launch_bounds__(128)
void attn_wmma(const _Float16* __restrict__ qh, const _Float16* __restrict__ kh,
               const _Float16* __restrict__ vT, _Float16* __restrict__ attn_h) {
  constexpr int L = 2048, NH = 14, NKV = 2, G = 7;
  int lane = threadIdx.x & 31;
  int wave = threadIdx.x >> 5;
  int gid  = blockIdx.x * (blockDim.x >> 5) + wave;   // over B*NH*(L/16) = 7168
  int qt = gid & 127;            // L/16 = 128
  int bh = gid >> 7;             // b*NH + h
  int b = bh / NH, h = bh % NH;
  int kvh = h / G;

  const _Float16* Q  = qh + (size_t)bh * L * 64;
  const _Float16* Kp = kh + (size_t)(b * NKV + kvh) * L * 64;
  const _Float16* Vt = vT + (size_t)(b * NKV + kvh) * 64 * L;

  extern __shared__ _Float16 smem[];
  _Float16* pL = smem + wave * (16 * 32);   // private 1KB P-staging slice per wave

  int m   = lane & 15;
  int hi  = lane >> 4;
  int hi8 = hi << 3;

  const _Float16* qrowp = Q + (size_t)(qt * 16 + m) * 64;
  v16h qa0 = ld_frag(qrowp, hi8);        // K-dim chunk 0..31
  v16h qa1 = ld_frag(qrowp + 32, hi8);   // K-dim chunk 32..63

  float mrow[8], lrow[8];
#pragma unroll
  for (int r = 0; r < 8; ++r) { mrow[r] = -3.0e38f; lrow[r] = 0.0f; }
  v8f o[4] = {};

  const float scale = 0.125f;            // 1/sqrt(64)
  int qmax = qt * 16 + 15;
  int nkb  = (qmax >> 5) + 1;            // only causal 32-key blocks

  for (int kb = 0; kb < nkb; ++kb) {
    int key0 = kb * 32;
    // ---- S = Q K^T for 32 keys (two 16x16 tiles, K=64 = 2 WMMA each)
    v8f s0 = {}, s1 = {};
    {
      const _Float16* kr0 = Kp + (size_t)(key0 + m) * 64;
      const _Float16* kr1 = Kp + (size_t)(key0 + 16 + m) * 64;
      v16h b00 = ld_frag(kr0, hi8), b01 = ld_frag(kr0 + 32, hi8);
      v16h b10 = ld_frag(kr1, hi8), b11 = ld_frag(kr1 + 32, hi8);
      s0 = wmma32(qa0, b00, s0); s0 = wmma32(qa1, b01, s0);
      s1 = wmma32(qa0, b10, s1); s1 = wmma32(qa1, b11, s1);
    }
    // ---- scale + causal mask + per-row max over this block
    float rmax[8];
#pragma unroll
    for (int r = 0; r < 8; ++r) {
      int qrow = qt * 16 + r + 8 * hi;
      float v0 = s0[r] * scale + ((key0 + m)      > qrow ? -1.0e9f : 0.0f);
      float v1 = s1[r] * scale + ((key0 + 16 + m) > qrow ? -1.0e9f : 0.0f);
      s0[r] = v0; s1[r] = v1;
      rmax[r] = fmaxf(v0, v1);
    }
#pragma unroll
    for (int r = 0; r < 8; ++r) {
      float v = rmax[r];
      v = fmaxf(v, __shfl_xor(v, 1, 16));
      v = fmaxf(v, __shfl_xor(v, 2, 16));
      v = fmaxf(v, __shfl_xor(v, 4, 16));
      v = fmaxf(v, __shfl_xor(v, 8, 16));
      rmax[r] = v;
    }
    // ---- online softmax update; stage P (f16) into LDS in row-major 16x32
    float psum[8];
#pragma unroll
    for (int r = 0; r < 8; ++r) {
      float mnew = fmaxf(mrow[r], rmax[r]);
      float corr = __expf(mrow[r] - mnew);
      mrow[r] = mnew;
      float p0 = __expf(s0[r] - mnew);
      float p1 = __expf(s1[r] - mnew);
#pragma unroll
      for (int c = 0; c < 4; ++c) o[c][r] *= corr;
      lrow[r] *= corr;
      psum[r] = p0 + p1;
      pL[(r + 8 * hi) * 32 + m]      = (_Float16)p0;
      pL[(r + 8 * hi) * 32 + 16 + m] = (_Float16)p1;
    }
#pragma unroll
    for (int r = 0; r < 8; ++r) {
      float v = psum[r];
      v += __shfl_xor(v, 1, 16);
      v += __shfl_xor(v, 2, 16);
      v += __shfl_xor(v, 4, 16);
      v += __shfl_xor(v, 8, 16);
      lrow[r] += v;
    }
    // DS ops are in-order within a wave; wait + clobber stops compiler reordering
    asm volatile("s_wait_dscnt 0x0" ::: "memory");
    // ---- reload P as A-fragment (C-layout -> A-layout transpose via LDS)
    v16h pa = ld_frag(pL + m * 32, hi8);
    // ---- O += P V  (V from vT rows: contiguous over key dim)
#pragma unroll
    for (int c = 0; c < 4; ++c) {
      const _Float16* vrow = Vt + (size_t)(c * 16 + m) * L + key0;
      v16h bv = ld_frag(vrow, hi8);
      o[c] = wmma32(pa, bv, o[c]);
    }
  }
  // ---- normalize and emit f16 for the output projection
#pragma unroll
  for (int c = 0; c < 4; ++c) {
#pragma unroll
    for (int r = 0; r < 8; ++r) {
      size_t row = (size_t)b * L + qt * 16 + r + 8 * hi;
      int col = h * 64 + c * 16 + m;
      attn_h[row * (NH * 64) + col] = (_Float16)(o[c][r] / lrow[r]);
    }
  }
}

// ---------------------------------------------------------------- launch
extern "C" void kernel_launch(void* const* d_in, const int* in_sizes, int n_in,
                              void* d_out, int out_size, void* d_ws, size_t ws_size,
                              hipStream_t stream) {
  constexpr int Bc = 4, Lc = 2048, HIDc = 896, NHc = 14, NKVc = 2, HDc = 64;
  const size_t Mrows = (size_t)Bc * Lc;           // 8192

  const float* x    = (const float*)d_in[0];
  const float* cosp = (const float*)d_in[1];
  const float* sinp = (const float*)d_in[2];
  // d_in[3] = mask: unused, causal+(-1e9) computed analytically
  const float* wq = (const float*)d_in[4];
  const float* bq = (const float*)d_in[5];
  const float* wk = (const float*)d_in[6];
  const float* bk = (const float*)d_in[7];
  const float* wv = (const float*)d_in[8];
  const float* bv = (const float*)d_in[9];
  const float* wo = (const float*)d_in[10];

  char* ws = (char*)d_ws;
  size_t off = 0;
  auto alloc = [&](size_t bytes) -> char* {
    char* p = ws + off;
    off += (bytes + 255) & ~(size_t)255;
    return p;
  };
  _Float16* xh  = (_Float16*)alloc(Mrows * HIDc * 2);
  _Float16* wqh = (_Float16*)alloc((size_t)NHc * HDc * HIDc * 2);
  _Float16* wkh = (_Float16*)alloc((size_t)NKVc * HDc * HIDc * 2);
  _Float16* wvh = (_Float16*)alloc((size_t)NKVc * HDc * HIDc * 2);
  _Float16* woh = (_Float16*)alloc((size_t)HIDc * NHc * HDc * 2);
  float* qf = (float*)alloc(Mrows * NHc * HDc * 4);
  float* kf = (float*)alloc(Mrows * NKVc * HDc * 4);
  float* vf = (float*)alloc(Mrows * NKVc * HDc * 4);
  _Float16* qhb = (_Float16*)alloc(Mrows * NHc * HDc * 2);
  _Float16* khb = (_Float16*)alloc(Mrows * NKVc * HDc * 2);
  _Float16* vtb = (_Float16*)alloc(Mrows * NKVc * HDc * 2);
  _Float16* ath = (_Float16*)alloc(Mrows * NHc * HDc * 2);

  auto cvt = [&](const float* in, _Float16* out, size_t n) {
    int thr = 256;
    int blk = (int)((n + thr - 1) / thr);
    cvt_f32_f16<<<blk, thr, 0, stream>>>(in, out, n);
  };
  cvt(x,  xh,  Mrows * HIDc);
  cvt(wq, wqh, (size_t)NHc * HDc * HIDc);
  cvt(wk, wkh, (size_t)NKVc * HDc * HIDc);
  cvt(wv, wvh, (size_t)NKVc * HDc * HIDc);
  cvt(wo, woh, (size_t)HIDc * NHc * HDc);

  auto gemm = [&](const _Float16* A, const _Float16* W, const float* bias,
                  float* C, int M, int N, int K) {
    int tiles = (M >> 5) * (N >> 6);
    int blocks = (tiles + 3) / 4;                // 4 waves / block
    gemm_xwt_wmma<<<blocks, 128, 0, stream>>>(A, W, bias, C, M, N, K);
  };
  gemm(xh, wqh, bq, qf, (int)Mrows, NHc * HDc, HIDc);    // Q: 8192x896
  gemm(xh, wkh, bk, kf, (int)Mrows, NKVc * HDc, HIDc);   // K: 8192x128
  gemm(xh, wvh, bv, vf, (int)Mrows, NKVc * HDc, HIDc);   // V: 8192x128

  {
    long long nq = (long long)Bc * Lc * NHc * 64;
    rope_pack<<<(int)((nq + 255) / 256), 256, 0, stream>>>(qf, cosp, sinp, qhb, NHc, 1);
    long long nk = (long long)Bc * Lc * NKVc * 64;
    rope_pack<<<(int)((nk + 255) / 256), 256, 0, stream>>>(kf, cosp, sinp, khb, NKVc, 1);
    long long nv = (long long)Bc * Lc * NKVc * 64;
    pack_vT<<<(int)((nv + 255) / 256), 256, 0, stream>>>(vf, vtb);
  }

  {
    int totalWaves = Bc * NHc * (Lc / 16);       // 7168
    int blocks = totalWaves / 4;                 // 1792, exact
    size_t shmem = 4 * 16 * 32 * sizeof(_Float16);  // 4 KB (1 KB per wave)
    attn_wmma<<<blocks, 128, shmem, stream>>>(qhb, khb, vtb, ath);
  }

  gemm(ath, woh, nullptr, (float*)d_out, (int)Mrows, HIDc, NHc * HDc);
  (void)in_sizes; (void)n_in; (void)out_size; (void)ws_size;
}
